// HighOrderBilinearFusion_84224308674766
// MI455X (gfx1250) — matrix operand order repaired and checked by reading, not compiled, not run on hardware
//
#include <hip/hip_runtime.h>

typedef __attribute__((ext_vector_type(16))) __bf16 v16bf;
typedef __attribute__((ext_vector_type(8)))  float  v8f;
typedef unsigned int u32x4 __attribute__((ext_vector_type(4)));
typedef int          i32x4 __attribute__((ext_vector_type(4)));
typedef int          i32x8 __attribute__((ext_vector_type(8)));

#define NPIX  25088   // 8*56*56
#define HW    3136    // 56*56
#define CIN   128
#define MID   64
#define OUTC  256
#define KDIM  4096    // MID*MID
#define WP_PITCH 36   // dwords per staged Wp row (32 data + 4 pad; 16B-aligned, bank-spread)
#define P1_PITCH 65   // floats per staged p1 row (conflict-free scalar reads)

#if defined(__gfx1250__) && __has_builtin(__builtin_amdgcn_tensor_load_to_lds)
#define USE_TDM 1
#else
#define USE_TDM 0
#endif

// ---------------- kernel 0: Wp f32 -> bf16 (L2-resident weight image) ----------------
__global__ void wp_to_bf16(const float* __restrict__ wp, __bf16* __restrict__ out) {
  int i = blockIdx.x * 256 + threadIdx.x;
  if (i < OUTC * KDIM) out[i] = (__bf16)wp[i];
}

// ---------------- kernel 1: 1x1-conv projections p1,p2 (f32 VALU, tiny FLOP share) ----
__global__ __launch_bounds__(256) void proj_kernel(
    const float* __restrict__ x1, const float* __restrict__ x2,
    const float* __restrict__ W1, const float* __restrict__ W2,
    float* __restrict__ p1f, float* __restrict__ p2f) {
  __shared__ float xs1[32 * 129];
  __shared__ float xs2[32 * 129];
  const int tid = threadIdx.x;
  const int px  = tid & 31;        // pixel within tile (lane)
  const int grp = tid >> 5;        // 0..7, wave-uniform -> W loads are scalar
  const int n0  = blockIdx.x * 32; // 3136 % 32 == 0 -> tile never crosses batch
  const int b   = n0 / HW;
  const int hw0 = n0 - b * HW;

  #pragma unroll
  for (int k = 0; k < 16; ++k) {
    int c = grp + 8 * k;
    size_t gidx = ((size_t)(b * CIN + c)) * HW + hw0 + px;  // coalesced over px
    xs1[px * 129 + c] = x1[gidx];
    xs2[px * 129 + c] = x2[gidx];
  }
  __syncthreads();

  float a1[8], a2[8];
  #pragma unroll
  for (int m = 0; m < 8; ++m) { a1[m] = 0.f; a2[m] = 0.f; }
  const int m0 = grp * 8;
  for (int c = 0; c < CIN; ++c) {
    float xv1 = xs1[px * 129 + c];   // pitch 129 -> conflict-free
    float xv2 = xs2[px * 129 + c];
    #pragma unroll
    for (int m = 0; m < 8; ++m) {
      a1[m] = fmaf(xv1, W1[(m0 + m) * CIN + c], a1[m]);
      a2[m] = fmaf(xv2, W2[(m0 + m) * CIN + c], a2[m]);
    }
  }
  const int n = n0 + px;
  float4* d1 = (float4*)(p1f + (size_t)n * MID + m0);
  float4* d2 = (float4*)(p2f + (size_t)n * MID + m0);
  d1[0] = make_float4(a1[0], a1[1], a1[2], a1[3]);
  d1[1] = make_float4(a1[4], a1[5], a1[6], a1[7]);
  d2[0] = make_float4(a2[0], a2[1], a2[2], a2[3]);
  d2[1] = make_float4(a2[4], a2[5], a2[6], a2[7]);
}

#if USE_TDM
// Issue one TDM 2D tile load: 256 rows x 32 dwords (Wp[:, c*64 .. +64) in bf16),
// row stride 2048 dwords, with HW padding of 4 dwords after every 32 dwords so the
// LDS image lands at WP_PITCH=36 (bank-friendly, 16B-aligned rows).
__device__ __forceinline__ void tdm_load_wp_chunk(unsigned lds_off, const void* gsrc) {
  unsigned long long ga = (unsigned long long)(size_t)gsrc;
  // D# group0: count=1 | lds_addr | global_addr[56:0] | type=2 ("image")
  u32x4 g0 = { 1u,
               lds_off,
               (unsigned)ga,
               (unsigned)((ga >> 32) & 0x1FFFFFFu) | (2u << 30) };
  // D# group1: data_size=4B, pad_enable, pad_interval=4 (32 dwords), pad_amount=3 (4 dwords)
  // tensor_dim0=2048 (dwords/row), tensor_dim1=256, tile_dim0=32, tile_dim1=256,
  // tensor_dim0_stride=2048 dwords.
  i32x8 g1 = { (int)((2u << 16) | (1u << 20) | (4u << 22) | (3u << 25)),
               (int)(2048u << 16),   // ab_addr=0 | tensor_dim0[15:0]
               (int)(256u << 16),    // tensor_dim0[31:16]=0 | tensor_dim1[15:0]
               (int)(32u << 16),     // tensor_dim1[31:16]=0 | tile_dim0=32
               (int)256,             // tile_dim1=256 | tile_dim2=0
               (int)2048,            // tensor_dim0_stride[31:0]
               0, 0 };               // stride hi / tensor_dim1_stride = 0 (2D)
  i32x4 gz = { 0, 0, 0, 0 };
#if __clang_major__ >= 23
  i32x8 gz8 = { 0, 0, 0, 0, 0, 0, 0, 0 };
  __builtin_amdgcn_tensor_load_to_lds(g0, g1, gz, gz, gz8, 0);
#else
  __builtin_amdgcn_tensor_load_to_lds(g0, g1, gz, gz, 0);
#endif
}
#endif

// ---------------- kernel 2: bilinear GEMM (bf16 WMMA) + fused LayerNorm + GELU -------
// out[o, n] = sum_k Wp[o, k] * (p1[n, k>>6] * p2[n, k&63]),  M=256, K=4096, N=25088
// One wave: 16 pixels x all 256 outputs (16 accumulator tiles). B-fragment built once
// per 32-K slab, reused by 16 WMMAs. Wp K-chunks are double-buffered in LDS via the
// Tensor Data Mover (chunk c+1 streams in while chunk c is consumed). Epilogue does
// LayerNorm fully in-wave (one shfl_xor pair) + exact GELU, stores straight to HBM.
__global__ __launch_bounds__(128, 1) void bilinear_wmma_kernel(
    const float* __restrict__ p1f, const float* __restrict__ p2f,
    const unsigned int* __restrict__ wpbf,   // Wp bf16, [o][k] row-major, 2048 dwords/row
    const float* __restrict__ gamma, const float* __restrict__ beta,
    float* __restrict__ out) {
  __shared__ float p1s[64 * P1_PITCH];
  __shared__ unsigned int wpc[2][OUTC * WP_PITCH];
  __shared__ float gs[OUTC];
  __shared__ float bs[OUTC];

  const int tid  = threadIdx.x;
  const int lane = tid & 31;
  const int wv   = tid >> 5;     // wave in block: owns pixel tile
  const int ln   = lane & 15;    // pixel column / o-row within fragment
  const int half = lane >> 4;    // K-half selector per WMMA layouts
  const int n0b  = blockIdx.x * 64;
  const int n    = n0b + wv * 16 + ln;

  // Stage p1 (f32) for the block's 64 pixels; stage gamma/beta.
  #pragma unroll
  for (int r = 0; r < 32; ++r) {
    int flat = tid + 128 * r;    // coalesced read of p1f[n0b*64 .. +4096)
    p1s[(flat >> 6) * P1_PITCH + (flat & 63)] = p1f[(size_t)n0b * MID + flat];
  }
  gs[tid] = gamma[tid]; gs[tid + 128] = gamma[tid + 128];
  bs[tid] = beta[tid];  bs[tid + 128] = beta[tid + 128];

  // Per-lane register copy of exactly the p2 values this lane's B-fragments need:
  // p2r[16*dh + j] = p2[n][32*dh + 16*half + j]
  float p2r[32];
  {
    const float4* p2p = (const float4*)(p2f + (size_t)n * MID);
    #pragma unroll
    for (int q = 0; q < 4; ++q) {
      float4 v = p2p[4 * half + q];
      p2r[4*q+0] = v.x; p2r[4*q+1] = v.y; p2r[4*q+2] = v.z; p2r[4*q+3] = v.w;
    }
    #pragma unroll
    for (int q = 0; q < 4; ++q) {
      float4 v = p2p[8 + 4 * half + q];
      p2r[16+4*q+0] = v.x; p2r[16+4*q+1] = v.y; p2r[16+4*q+2] = v.z; p2r[16+4*q+3] = v.w;
    }
  }

  v8f zero;
  #pragma unroll
  for (int r = 0; r < 8; ++r) zero[r] = 0.0f;
  v8f acc[16];
  #pragma unroll
  for (int t = 0; t < 16; ++t) acc[t] = zero;

#if USE_TDM
  if (wv == 0)   // uniform branch: TDM issues once per workgroup
    tdm_load_wp_chunk((unsigned)(size_t)(void*)&wpc[0][0], wpbf);
#endif

  for (int c = 0; c < MID; ++c) {
    const unsigned int* wcur = &wpc[c & 1][0];
#if USE_TDM
    __builtin_amdgcn_s_wait_tensorcnt(0);  // wave0: chunk c landed; others: no-op
    __syncthreads();                       // chunk c visible; buf[(c+1)&1] free
    if (wv == 0 && (c + 1) < MID)
      tdm_load_wp_chunk((unsigned)(size_t)(void*)&wpc[(c + 1) & 1][0],
                        (const unsigned char*)wpbf + (size_t)(c + 1) * 128);
#else
    __syncthreads();  // previous chunk fully consumed
    #pragma unroll
    for (int rr = 0; rr < 2; ++rr) {
      int o = tid + 128 * rr;
      const uint4* src = (const uint4*)((const unsigned char*)wpbf + (size_t)o * (KDIM * 2) + c * 128);
      uint4* dst = (uint4*)const_cast<unsigned int*>(&wpc[c & 1][o * WP_PITCH]);
      #pragma unroll
      for (int q = 0; q < 8; ++q) dst[q] = src[q];
    }
    __syncthreads();  // chunk visible
#endif

    const float p1v = p1s[(wv * 16 + ln) * P1_PITCH + c];
    #pragma unroll
    for (int dh = 0; dh < 2; ++dh) {    // K-slab k0 = c*64 + 32*dh
      // B fragment (32x16 bf16): lane holds v[k0 + 16*half + j, n], j=0..15
      v16bf bmat;
      #pragma unroll
      for (int j = 0; j < 16; ++j) bmat[j] = (__bf16)(p1v * p2r[16 * dh + j]);
      #pragma unroll
      for (int t = 0; t < 16; ++t) {
        // A fragment (16x32 bf16): lane = row o=16t+ln; 16 contiguous K selected by half
        const unsigned int* rp = wcur + (16 * t + ln) * WP_PITCH;
        union { v16bf v; uint4 q[2]; } ua;
        ua.q[0] = *(const uint4*)(rp + 16 * dh + 8 * half);
        ua.q[1] = *(const uint4*)(rp + 16 * dh + 8 * half + 4);
        acc[t] = __builtin_amdgcn_wmma_f32_16x16x32_bf16(
            false, ua.v, false, bmat, (short)0, acc[t], false, false);
      }
    }
  }
  __syncthreads();

  // ---- fused LayerNorm (over 256 channels, in-wave) + exact GELU + store ----
  float s1 = 0.f, s2 = 0.f;
  #pragma unroll
  for (int t = 0; t < 16; ++t)
    #pragma unroll
    for (int r = 0; r < 8; ++r) { float v = acc[t][r]; s1 += v; s2 += v * v; }
  s1 += __shfl_xor(s1, 16, 32);  // combine lane pair across halves (same pixel n)
  s2 += __shfl_xor(s2, 16, 32);
  const float mean = s1 * (1.0f / OUTC);
  const float var  = s2 * (1.0f / OUTC) - mean * mean;
  const float rstd = rsqrtf(var + 1e-5f);

  const int b  = n / HW;
  const int hw = n - b * HW;
  float* obase = out + ((size_t)b * OUTC) * HW + hw;
  #pragma unroll
  for (int t = 0; t < 16; ++t) {
    #pragma unroll
    for (int r = 0; r < 8; ++r) {
      const int o = 16 * t + r + 8 * half;   // D-matrix layout
      float v = (acc[t][r] - mean) * rstd * gs[o] + bs[o];
      v = 0.5f * v * (1.0f + erff(v * 0.70710678118654752f));
      obase[(size_t)o * HW] = v;             // contiguous over lanes 0..15
    }
  }
}

// ---------------- launcher ----------------
extern "C" void kernel_launch(void* const* d_in, const int* in_sizes, int n_in,
                              void* d_out, int out_size, void* d_ws, size_t ws_size,
                              hipStream_t stream) {
  const float* x1    = (const float*)d_in[0];
  const float* x2    = (const float*)d_in[1];
  const float* W1    = (const float*)d_in[2];
  const float* W2    = (const float*)d_in[3];
  const float* Wp    = (const float*)d_in[4];
  const float* gamma = (const float*)d_in[5];
  const float* beta  = (const float*)d_in[6];

  // workspace: p1f (6.42MB) | p2f (6.42MB) | Wp bf16 (2MB)  => 14.3MB total
  float*  p1f  = (float*)d_ws;
  float*  p2f  = p1f + (size_t)NPIX * MID;
  __bf16* wpbf = (__bf16*)(p2f + (size_t)NPIX * MID);

  wp_to_bf16<<<(OUTC * KDIM) / 256, 256, 0, stream>>>(Wp, wpbf);
  proj_kernel<<<NPIX / 32, 256, 0, stream>>>(x1, x2, W1, W2, p1f, p2f);
  bilinear_wmma_kernel<<<NPIX / 64, 128, 0, stream>>>(
      p1f, p2f, (const unsigned int*)wpbf, gamma, beta, (float*)d_out);

  (void)in_sizes; (void)n_in; (void)out_size; (void)ws_size;
}
